// naive_nn_21938692948259
// MI455X (gfx1250) — matrix operand order repaired and checked by reading, not compiled, not run on hardware
//
#include <hip/hip_runtime.h>
#include <hip/hip_bf16.h>
#include <math.h>

// ---------- types ----------
typedef __attribute__((ext_vector_type(16))) __bf16 v16bf;
typedef __attribute__((ext_vector_type(8)))  __bf16 v8bf;
typedef __attribute__((ext_vector_type(8)))  float  v8f;

// ---------- problem constants ----------
#define BATCH   2048
#define IN_DIM  60
#define TTL     30
#define HDIM    512
#define NLAY    5
#define OUTDIM  23
#define T_PTS   90
#define SUBN    10

#define P_D_C 0.2f
#define P_H_C 0.15f
#define P_V_C 0.25f
#define LN2_C 0.6931471805599453f
#define R_I_C  (LN2_C / 5.0f)
#define R_D_C  (LN2_C / 2.0f)
#define R_RI_C (LN2_C / 10.0f)
#define R_RH_C (LN2_C / 15.0f)
#define R_RV_C (LN2_C / 10.0f)
#define PI_C 3.14159265358979323846f

// output layout (floats)
#define SOL_ELEMS  (BATCH * T_PTS * 16)          // 2949120
#define DNN_ELEMS  (BATCH * OUTDIM)              // 47104
#define TSO_ELEMS  (BATCH * HDIM)                // 1048576

// ---------- fragment helpers (CDNA5 WMMA bf16 16x16x32 layouts) ----------
// A (16x32, 16-bit): lane L: m = L&15, half = L>>4.
//   elems 0..7  : K = k0 + half*8 + i       (contiguous 16B)
//   elems 8..15 : K = k0 + 16 + half*8 + i  (contiguous 16B)
__device__ inline v16bf load_afrag(const __bf16* __restrict__ X, int ldx,
                                   int row, int k0, int half) {
  const __bf16* p = X + (size_t)row * ldx + k0 + half * 8;
  v8bf lo = *(const v8bf*)(p);
  v8bf hi = *(const v8bf*)(p + 16);
  v16bf a;
#pragma unroll
  for (int i = 0; i < 8; ++i) { a[i] = lo[i]; a[i + 8] = hi[i]; }
  return a;
}

// B (32x16, 16-bit), B[k][n] = W[n][k]: lane L: n = L&15, half = L>>4.
//   elems 0..7 : K = k0 + half*8 + j; elems 8..15 : K = k0 + 16 + half*8 + (j-8)
__device__ inline v16bf load_bfrag(const __bf16* __restrict__ W, int ldw,
                                   int rowN, int k0, int half) {
  const __bf16* p = W + (size_t)rowN * ldw + k0 + half * 8;
  v8bf b0 = *(const v8bf*)(p);
  v8bf b1 = *(const v8bf*)(p + 16);
  v16bf b;
#pragma unroll
  for (int i = 0; i < 8; ++i) { b[i] = b0[i]; b[i + 8] = b1[i]; }
  return b;
}

__device__ inline v16bf zero16() {
  v16bf z;
#pragma unroll
  for (int i = 0; i < 16; ++i) z[i] = (__bf16)0.0f;
  return z;
}

__device__ inline float sigf(float x) { return 1.0f / (1.0f + expf(-x)); }

// ---------- conversion / prep kernels ----------
__global__ void cvt_pad_kernel(const float* __restrict__ src, __bf16* __restrict__ dst,
                               int rows, int cs, int cd) {
  int i = blockIdx.x * blockDim.x + threadIdx.x;
  int total = rows * cd;
  if (i >= total) return;
  int r = i / cd, c = i - r * cd;
  dst[i] = (c < cs) ? (__bf16)src[(size_t)r * cs + c] : (__bf16)0.0f;
}

__global__ void prep_x0_kernel(const float* __restrict__ mi, const float* __restrict__ pop,
                               __bf16* __restrict__ x0, __bf16* __restrict__ metain) {
  int i = blockIdx.x * blockDim.x + threadIdx.x;
  if (i >= BATCH * 32) return;
  int b = i >> 5, c = i & 31;
  float invp = 1.0f / pop[b];
  x0[i]     = (c < TTL) ? (__bf16)(mi[(size_t)b * IN_DIM + c] * invp) : (__bf16)0.0f;
  metain[i] = (c < TTL) ? (__bf16)(mi[(size_t)b * IN_DIM + TTL + c]) : (__bf16)0.0f;
}

// ---------- fused LSTM layer: one wave = 32 rows x 16 cols of x_next ----------
// Register-blocked 2x in M: B fragments (gate weights) are reused by two A
// fragments -> 1.5 VMEM loads per WMMA instead of 2.5.
__global__ __launch_bounds__(32)
void lstm_layer_kernel(const __bf16* __restrict__ X, int ldx, int Kx,
                       const __bf16* __restrict__ H,              // (B,512) bf16
                       const __bf16* __restrict__ Wi, int ldwi,   // (2048,Kx)
                       const __bf16* __restrict__ Wh,             // (2048,512)
                       const float* __restrict__ bi, const float* __restrict__ bh,
                       const float* __restrict__ c0,              // (B,512) f32
                       __bf16* __restrict__ Xout, int ldxo,
                       float* __restrict__ tso) {
  const int m0 = blockIdx.x * 32;
  const int n0 = blockIdx.y * 16;
  const int lane = threadIdx.x;
  const int half = lane >> 4;
  const int mr = lane & 15;
  const int nc = lane & 15;
  const int rowA0 = m0 + mr;
  const int rowA1 = m0 + 16 + mr;

  v8f acc[2][4];
#pragma unroll
  for (int t = 0; t < 2; ++t)
#pragma unroll
    for (int g = 0; g < 4; ++g)
#pragma unroll
      for (int r = 0; r < 8; ++r) acc[t][g][r] = 0.0f;

  // input contribution: x @ Wi.T
  for (int k0 = 0; k0 < Kx; k0 += 32) {
    v16bf a0 = load_afrag(X, ldx, rowA0, k0, half);
    v16bf a1 = load_afrag(X, ldx, rowA1, k0, half);
#pragma unroll
    for (int g = 0; g < 4; ++g) {
      v16bf b = load_bfrag(Wi, ldwi, g * HDIM + n0 + nc, k0, half);
      acc[0][g] = __builtin_amdgcn_wmma_f32_16x16x32_bf16(
          false, a0, false, b, (short)0, acc[0][g], false, false);
      acc[1][g] = __builtin_amdgcn_wmma_f32_16x16x32_bf16(
          false, a1, false, b, (short)0, acc[1][g], false, false);
    }
  }
  // hidden contribution: h0 @ Whh.T
  for (int k0 = 0; k0 < HDIM; k0 += 32) {
    v16bf a0 = load_afrag(H, HDIM, rowA0, k0, half);
    v16bf a1 = load_afrag(H, HDIM, rowA1, k0, half);
#pragma unroll
    for (int g = 0; g < 4; ++g) {
      v16bf b = load_bfrag(Wh, HDIM, g * HDIM + n0 + nc, k0, half);
      acc[0][g] = __builtin_amdgcn_wmma_f32_16x16x32_bf16(
          false, a0, false, b, (short)0, acc[0][g], false, false);
      acc[1][g] = __builtin_amdgcn_wmma_f32_16x16x32_bf16(
          false, a1, false, b, (short)0, acc[1][g], false, false);
    }
  }

  float bias[4];
#pragma unroll
  for (int g = 0; g < 4; ++g)
    bias[g] = bi[g * HDIM + n0 + nc] + bh[g * HDIM + n0 + nc];

  const int col = n0 + nc;
#pragma unroll
  for (int t = 0; t < 2; ++t) {
#pragma unroll
    for (int r = 0; r < 8; ++r) {
      int m = m0 + t * 16 + 8 * half + r;
      float gi = acc[t][0][r] + bias[0];
      float gf = acc[t][1][r] + bias[1];
      float gg = acc[t][2][r] + bias[2];
      float go = acc[t][3][r] + bias[3];
      float cprev = c0[(size_t)m * HDIM + col];
      float cnew  = sigf(gf) * cprev + sigf(gi) * tanhf(gg);
      float xn    = sigf(go) * tanhf(cnew);
      Xout[(size_t)m * ldxo + col] = (__bf16)xn;
      if (tso) tso[(size_t)m * HDIM + col] = xn;
    }
  }
}

// ---------- generic GEMM: out = act(A @ W.T + bias), A:(M,K) W:(N,K) ----------
// 2x2 register blocking: one wave = 32x32 output tile, 2 loads per WMMA.
// act: 0 = none, 1 = relu, 2 = tanh
__global__ __launch_bounds__(32)
void gemm_bias_act_kernel(const __bf16* __restrict__ A, int lda,
                          const __bf16* __restrict__ W, int ldw,
                          const float* __restrict__ bias,
                          float* __restrict__ Cf, __bf16* __restrict__ Cb, int ldc,
                          int M, int N, int K, int act) {
  const int m0 = blockIdx.x * 32;
  const int n0 = blockIdx.y * 32;
  const int lane = threadIdx.x;
  const int half = lane >> 4;
  const int mr = lane & 15;
  const int nc = lane & 15;
  const int rowA0 = m0 + mr;
  const int rowA1 = m0 + 16 + mr;
  const int nw0 = n0 + nc;
  const int nw1 = n0 + 16 + nc;

  v8f acc[2][2];
#pragma unroll
  for (int t = 0; t < 2; ++t)
#pragma unroll
    for (int u = 0; u < 2; ++u)
#pragma unroll
      for (int r = 0; r < 8; ++r) acc[t][u][r] = 0.0f;

  for (int k0 = 0; k0 < K; k0 += 32) {
    v16bf a0 = load_afrag(A, lda, rowA0, k0, half);
    v16bf a1 = load_afrag(A, lda, rowA1, k0, half);
    v16bf b0 = (nw0 < N) ? load_bfrag(W, ldw, nw0, k0, half) : zero16();
    v16bf b1 = (nw1 < N) ? load_bfrag(W, ldw, nw1, k0, half) : zero16();
    acc[0][0] = __builtin_amdgcn_wmma_f32_16x16x32_bf16(
        false, a0, false, b0, (short)0, acc[0][0], false, false);
    acc[0][1] = __builtin_amdgcn_wmma_f32_16x16x32_bf16(
        false, a0, false, b1, (short)0, acc[0][1], false, false);
    acc[1][0] = __builtin_amdgcn_wmma_f32_16x16x32_bf16(
        false, a1, false, b0, (short)0, acc[1][0], false, false);
    acc[1][1] = __builtin_amdgcn_wmma_f32_16x16x32_bf16(
        false, a1, false, b1, (short)0, acc[1][1], false, false);
  }

#pragma unroll
  for (int u = 0; u < 2; ++u) {
    int nw = (u == 0) ? nw0 : nw1;
    if (nw >= N) continue;
    float bs = bias[nw];
#pragma unroll
    for (int t = 0; t < 2; ++t) {
#pragma unroll
      for (int r = 0; r < 8; ++r) {
        int m = m0 + t * 16 + 8 * half + r;
        float v = acc[t][u][r] + bs;
        if (act == 1) v = fmaxf(v, 0.0f);
        else if (act == 2) v = tanhf(v);
        if (Cf) Cf[(size_t)m * ldc + nw] = v;
        if (Cb) Cb[(size_t)m * ldc + nw] = (__bf16)v;
      }
    }
  }
}

// ---------- dnn = sigmoid(raw * scale) * range ----------
__global__ void dnn_kernel(const float* __restrict__ raw, const float* __restrict__ rng,
                           const int* __restrict__ scale_i, float* __restrict__ dnn) {
  int i = blockIdx.x * blockDim.x + threadIdx.x;
  if (i >= BATCH * OUTDIM) return;
  float s = (float)scale_i[0];
  dnn[i] = sigf(raw[i] * s) * rng[i % OUTDIM];
}

// ---------- ODE (RK4, 1 thread per batch row) ----------
struct OdeParams {
  float alpha, days, r_s, r_dth, p_dth, r_dthdecay, jump, t_jump, stdn, N;
};

__device__ inline void ode_deriv(const OdeParams& P, float t,
                                 const float* __restrict__ y, float* __restrict__ dy) {
  float S = y[0], E = y[1], I = y[2];
  float AR = y[3], DHR = y[4], DQR = y[5], AD = y[6], DHD = y[7], DQD = y[8];
  float DVR = y[12], DVD = y[13];
  float dtj = t - P.t_jump;
  float gam = (2.0f / PI_C) * atanf(-(t - P.days) / 20.0f * P.r_s) + 1.0f
            + P.jump * expf(-(dtj * dtj) / (2.0f * P.stdn * P.stdn));
  float pdt = (2.0f / PI_C) * (P.p_dth - 0.001f)
            * (atanf(-t * P.r_dthdecay / 20.0f) + PI_C * 0.5f) + 0.001f;
  float inf = P.alpha * gam * S * I / P.N;
  float det = R_D_C * I;
  dy[0]  = -inf;
  dy[1]  = inf - R_I_C * E;
  dy[2]  = R_I_C * E - det;
  dy[3]  = det * (1.0f - pdt) * (1.0f - P_D_C) - R_RI_C * AR;
  dy[4]  = det * (1.0f - pdt) * P_D_C * P_H_C - R_RH_C * DHR;
  dy[5]  = det * (1.0f - pdt) * P_D_C * (1.0f - P_H_C) - R_RI_C * DQR;
  dy[6]  = det * pdt * (1.0f - P_D_C) - P.r_dth * AD;
  dy[7]  = det * pdt * P_D_C * P_H_C - P.r_dth * DHD;
  dy[8]  = det * pdt * P_D_C * (1.0f - P_H_C) - P.r_dth * DQD;
  dy[9]  = R_RI_C * (AR + DQR) + R_RH_C * DHR;
  dy[10] = P.r_dth * (AD + DQD + DHD);
  dy[11] = det * P_D_C * P_H_C;
  dy[12] = det * (1.0f - pdt) * P_D_C * P_H_C * P_V_C - R_RV_C * DVR;
  dy[13] = det * pdt * P_D_C * P_H_C * P_V_C - P.r_dth * DVD;
  dy[14] = P.r_dth * (DHD + DQD);
  dy[15] = det * P_D_C;
}

__global__ __launch_bounds__(256)
void ode_rk4_kernel(const float* __restrict__ dnn, const float* __restrict__ pop,
                    const float* __restrict__ ccn, const float* __restrict__ mort,
                    float* __restrict__ sol) {
  int b = blockIdx.x * blockDim.x + threadIdx.x;
  if (b >= BATCH) return;

  const float* p = dnn + (size_t)b * OUTDIM + 11;
  OdeParams P;
  P.alpha = p[0]; P.days = p[1]; P.r_s = p[2]; P.r_dth = p[3]; P.p_dth = p[4];
  P.r_dthdecay = p[5];
  float k1p = p[6], k2p = p[7];
  P.jump = p[8]; P.t_jump = p[9]; P.stdn = p[10];
  P.N = pop[b];

  float PopI = ccn[b];
  float PopD = floorf(mort[b] * PopI);
  float R0   = (PopI - PopD > 5.0f * PopD) ? 5.0f * PopD : 0.0f;
  float PopCI = PopI - PopD - R0;
  float CIoP = PopCI / P_D_C;
  float pd   = P.p_dth;

  float y[16];
  y[0]  = P.N - CIoP * (k1p + k2p) - R0 / P_D_C - PopD / P_D_C;
  y[1]  = CIoP * k1p;
  y[2]  = CIoP * k2p;
  y[3]  = (CIoP - PopCI) * (1.0f - pd);
  y[4]  = PopCI * P_H_C * (1.0f - pd);
  y[5]  = PopCI * (1.0f - P_H_C) * (1.0f - pd);
  y[6]  = (CIoP - PopCI) * pd;
  y[7]  = PopCI * P_H_C * pd;
  y[8]  = PopCI * (1.0f - P_H_C) * pd;
  y[9]  = R0 / P_D_C;
  y[10] = PopD / P_D_C;
  y[11] = PopCI * P_H_C;
  y[12] = PopCI * P_H_C * P_V_C * (1.0f - pd);
  y[13] = PopCI * P_H_C * P_V_C * pd;
  y[14] = PopD;
  y[15] = PopI;

  float* srow = sol + (size_t)b * T_PTS * 16;
#pragma unroll
  for (int s = 0; s < 16; ++s) srow[s] = y[s];

  const float DT = 1.0f / (float)SUBN;
  float ka[16], kb[16], kc[16], kd[16], yt[16];

  for (int seg = 0; seg < T_PTS - 1; ++seg) {
    float t0 = (float)seg;
    for (int j = 0; j < SUBN; ++j) {
      float t = t0 + (float)j * DT;
      ode_deriv(P, t, y, ka);
#pragma unroll
      for (int s = 0; s < 16; ++s) yt[s] = y[s] + 0.5f * DT * ka[s];
      ode_deriv(P, t + 0.5f * DT, yt, kb);
#pragma unroll
      for (int s = 0; s < 16; ++s) yt[s] = y[s] + 0.5f * DT * kb[s];
      ode_deriv(P, t + 0.5f * DT, yt, kc);
#pragma unroll
      for (int s = 0; s < 16; ++s) yt[s] = y[s] + DT * kc[s];
      ode_deriv(P, t + DT, yt, kd);
#pragma unroll
      for (int s = 0; s < 16; ++s)
        y[s] += (DT / 6.0f) * (ka[s] + 2.0f * kb[s] + 2.0f * kc[s] + kd[s]);
    }
    float* orow = srow + (size_t)(seg + 1) * 16;
#pragma unroll
    for (int s = 0; s < 16; ++s) orow[s] = y[s];
  }
}

// ---------- host orchestration ----------
extern "C" void kernel_launch(void* const* d_in, const int* in_sizes, int n_in,
                              void* d_out, int out_size, void* d_ws, size_t ws_size,
                              hipStream_t stream) {
  const float* mi    = (const float*)d_in[0];
  const float* pop   = (const float*)d_in[1];
  const float* ccn   = (const float*)d_in[2];
  const float* mort  = (const float*)d_in[3];
  const float* h0    = (const float*)d_in[4];
  const float* c0    = (const float*)d_in[5];
  const float* W_ih0 = (const float*)d_in[6];
  const float* W_ihR = (const float*)d_in[7];
  const float* W_hh  = (const float*)d_in[8];
  const float* b_ih  = (const float*)d_in[9];
  const float* b_hh  = (const float*)d_in[10];
  const float* mW1   = (const float*)d_in[11];
  const float* mb1   = (const float*)d_in[12];
  const float* mW2   = (const float*)d_in[13];
  const float* mb2   = (const float*)d_in[14];
  const float* rW1   = (const float*)d_in[15];
  const float* rb1   = (const float*)d_in[16];
  const float* rW2   = (const float*)d_in[17];
  const float* rb2   = (const float*)d_in[18];
  const float* rng   = (const float*)d_in[19];
  const int*   sscal = (const int*)d_in[20];

  float* out   = (float*)d_out;
  float* solO  = out;
  float* dnnO  = out + SOL_ELEMS;
  float* tsoO  = out + SOL_ELEMS + DNN_ELEMS;
  float* rawO  = out + SOL_ELEMS + DNN_ELEMS + TSO_ELEMS;

  // bump allocator over workspace
  char* ws = (char*)d_ws;
  size_t off = 0;
  auto abf = [&](size_t elems) -> __bf16* {
    __bf16* ptr = (__bf16*)(ws + off);
    off += ((elems * sizeof(__bf16)) + 255) & ~(size_t)255;
    return ptr;
  };

  __bf16* wih0p = abf((size_t)4 * HDIM * 32);              // (2048,32) padded
  __bf16* wihRb = abf((size_t)(NLAY - 1) * 4 * HDIM * HDIM);
  __bf16* whhb  = abf((size_t)NLAY * 4 * HDIM * HDIM);
  __bf16* mW1p  = abf((size_t)HDIM * 32);                  // (512,32) padded
  __bf16* mW2b  = abf((size_t)HDIM * HDIM);
  __bf16* rW1b  = abf((size_t)2 * HDIM * 2 * HDIM);
  __bf16* rW2b  = abf((size_t)OUTDIM * 2 * HDIM);
  __bf16* h0b   = abf((size_t)NLAY * BATCH * HDIM);
  __bf16* x0b   = abf((size_t)BATCH * 32);
  __bf16* mib   = abf((size_t)BATCH * 32);
  __bf16* xa    = abf((size_t)BATCH * HDIM);
  __bf16* xb    = abf((size_t)BATCH * HDIM);
  __bf16* m1b   = abf((size_t)BATCH * HDIM);
  __bf16* zb    = abf((size_t)BATCH * 2 * HDIM);
  __bf16* ztb   = abf((size_t)BATCH * 2 * HDIM);
  (void)ws_size; (void)in_sizes; (void)n_in; (void)out_size;

  auto blocks = [](size_t n) { return dim3((unsigned)((n + 255) / 256)); };

  // ---- weight / state conversion (fp32 -> bf16, pad K 30 -> 32) ----
  cvt_pad_kernel<<<blocks((size_t)4 * HDIM * 32), 256, 0, stream>>>(W_ih0, wih0p, 4 * HDIM, TTL, 32);
  cvt_pad_kernel<<<blocks((size_t)(NLAY - 1) * 4 * HDIM * HDIM), 256, 0, stream>>>(W_ihR, wihRb, (NLAY - 1) * 4 * HDIM, HDIM, HDIM);
  cvt_pad_kernel<<<blocks((size_t)NLAY * 4 * HDIM * HDIM), 256, 0, stream>>>(W_hh, whhb, NLAY * 4 * HDIM, HDIM, HDIM);
  cvt_pad_kernel<<<blocks((size_t)HDIM * 32), 256, 0, stream>>>(mW1, mW1p, HDIM, TTL, 32);
  cvt_pad_kernel<<<blocks((size_t)HDIM * HDIM), 256, 0, stream>>>(mW2, mW2b, HDIM, HDIM, HDIM);
  cvt_pad_kernel<<<blocks((size_t)4 * HDIM * HDIM), 256, 0, stream>>>(rW1, rW1b, 2 * HDIM, 2 * HDIM, 2 * HDIM);
  cvt_pad_kernel<<<blocks((size_t)OUTDIM * 2 * HDIM), 256, 0, stream>>>(rW2, rW2b, OUTDIM, 2 * HDIM, 2 * HDIM);
  cvt_pad_kernel<<<blocks((size_t)NLAY * BATCH * HDIM), 256, 0, stream>>>(h0, h0b, NLAY * BATCH, HDIM, HDIM);
  prep_x0_kernel<<<blocks((size_t)BATCH * 32), 256, 0, stream>>>(mi, pop, x0b, mib);

  // ---- LSTM stack (fused WMMA GEMM + cell), 32x16x4-gate tiles/wave ----
  dim3 wblk(32);
  dim3 gL(BATCH / 32, HDIM / 16);
  // layer 0
  lstm_layer_kernel<<<gL, wblk, 0, stream>>>(
      x0b, 32, 32, h0b, wih0p, 32, whhb,
      b_ih, b_hh, c0, xa, HDIM, nullptr);
  // layers 1..4
  __bf16* bufs[2] = {xa, xb};
  int cur = 0;
  for (int l = 1; l < NLAY; ++l) {
    const __bf16* Xin = bufs[cur];
    __bf16* Xout = (l == NLAY - 1) ? zb : bufs[cur ^ 1];
    int ldxo     = (l == NLAY - 1) ? 2 * HDIM : HDIM;
    float* tso   = (l == NLAY - 1) ? tsoO : nullptr;
    lstm_layer_kernel<<<gL, wblk, 0, stream>>>(
        Xin, HDIM, HDIM,
        h0b + (size_t)l * BATCH * HDIM,
        wihRb + (size_t)(l - 1) * 4 * HDIM * HDIM, HDIM,
        whhb + (size_t)l * 4 * HDIM * HDIM,
        b_ih + (size_t)l * 4 * HDIM, b_hh + (size_t)l * 4 * HDIM,
        c0 + (size_t)l * BATCH * HDIM,
        Xout, ldxo, tso);
    cur ^= 1;
  }

  // ---- meta MLP -> z[:, 512:] (32x32 tiles/wave) ----
  gemm_bias_act_kernel<<<dim3(BATCH / 32, HDIM / 32), wblk, 0, stream>>>(
      mib, 32, mW1p, 32, mb1, nullptr, m1b, HDIM, BATCH, HDIM, 32, 1);
  gemm_bias_act_kernel<<<dim3(BATCH / 32, HDIM / 32), wblk, 0, stream>>>(
      m1b, HDIM, mW2b, HDIM, mb2, nullptr, zb + HDIM, 2 * HDIM, BATCH, HDIM, HDIM, 1);

  // ---- readout ----
  gemm_bias_act_kernel<<<dim3(BATCH / 32, 2 * HDIM / 32), wblk, 0, stream>>>(
      zb, 2 * HDIM, rW1b, 2 * HDIM, rb1, nullptr, ztb, 2 * HDIM, BATCH, 2 * HDIM, 2 * HDIM, 2);
  gemm_bias_act_kernel<<<dim3(BATCH / 32, (OUTDIM + 31) / 32), wblk, 0, stream>>>(
      ztb, 2 * HDIM, rW2b, 2 * HDIM, rb2, rawO, nullptr, OUTDIM, BATCH, OUTDIM, 2 * HDIM, 0);

  // ---- dnn = sigmoid(raw*scale) * range ----
  dnn_kernel<<<blocks((size_t)BATCH * OUTDIM), 256, 0, stream>>>(rawO, rng, sscal, dnnO);

  // ---- ODE RK4 ----
  ode_rk4_kernel<<<dim3(BATCH / 256), 256, 0, stream>>>(dnnO, pop, ccn, mort, solO);
}
